// GodAreaModel_5729486373548
// MI455X (gfx1250) — compile-verified, matrix operand order checked by reading, not run on hardware
//
#include <hip/hip_runtime.h>
#include <math.h>

// MI455X / gfx1250, wave32. bf16 WMMA (f32 accum) GEMMs; Weff cached as bf16
// (134 MB, fits 192 MB L2) so the 4 recurrent ticks re-read weights from L2.
// Wave tile 64x64 (4x4 WMMA accumulators) -> ~32 FLOP/byte register reuse.

typedef __bf16 bf16;
typedef __attribute__((ext_vector_type(16))) __bf16 v16bf;
typedef __attribute__((ext_vector_type(8)))  float  v8f;

#define AREAS 4
#define AN    2048
#define BATCHN 512
#define DIN   1024
#define NCLS  1024
#define NTOT  (AREAS * AN)   // 8192

// ---------------- elementwise prep kernels ----------------

__global__ void cvt_f32_bf16(const float* __restrict__ in, bf16* __restrict__ out, int n) {
    int i = (blockIdx.x * blockDim.x + threadIdx.x) * 2;
    if (i < n) {
        out[i] = (bf16)in[i];
        if (i + 1 < n) out[i + 1] = (bf16)in[i + 1];
    }
}

__global__ void make_weff(const float* __restrict__ W, const float* __restrict__ mask,
                          bf16* __restrict__ out, int n) {
    int i = (blockIdx.x * blockDim.x + threadIdx.x) * 2;
    if (i < n) {
        float m0 = mask[i];     m0 = m0 < 0.f ? 0.f : (m0 > 1.f ? 1.f : m0);
        out[i] = (bf16)(W[i] * m0);
        if (i + 1 < n) {
            float m1 = mask[i + 1]; m1 = m1 < 0.f ? 0.f : (m1 > 1.f ? 1.f : m1);
            out[i + 1] = (bf16)(W[i + 1] * m1);
        }
    }
}

// Owp[c, j] = Ow[c, area_idx[j]]  (turns output scatter into a plain NT GEMM)
__global__ void make_owp(const float* __restrict__ Ow, const int* __restrict__ aidx,
                         bf16* __restrict__ owp) {
    int j = blockIdx.x * blockDim.x + threadIdx.x;   // 0..NTOT-1
    int c = blockIdx.y;                              // 0..NCLS-1
    owp[(size_t)c * NTOT + j] = (bf16)Ow[(size_t)c * NTOT + aidx[j]];
}

// ---------------- gate (data-dependent skip) ----------------

__global__ void zero_acc(float* acc) {
    if (threadIdx.x < AREAS) acc[threadIdx.x] = 0.f;
}

__global__ void __launch_bounds__(256) abs_reduce(const float* __restrict__ Z,
                                                  float* __restrict__ acc) {
    __shared__ float s[256];
    const int a = blockIdx.y;
    const size_t per = (size_t)BATCHN * AN;          // 1M elements per area
    const float* p = Z + (size_t)a * per;
    float sum = 0.f;
    for (size_t i = (size_t)blockIdx.x * 256 + threadIdx.x; i < per; i += 256 * 256)
        sum += fabsf(p[i]);
    s[threadIdx.x] = sum;
    __syncthreads();
    for (int st = 128; st > 0; st >>= 1) {
        if ((int)threadIdx.x < st) s[threadIdx.x] += s[threadIdx.x + st];
        __syncthreads();
    }
    if (threadIdx.x == 0) atomicAdd(&acc[a], s[0]);
}

__global__ void threshold_gate(const float* __restrict__ acc, float* __restrict__ gate) {
    int a = threadIdx.x;
    if (a < AREAS)
        gate[a] = (acc[a] * (1.0f / ((float)BATCHN * (float)AN)) > 0.05f) ? 1.f : 0.f;
}

// ---------------- WMMA GEMM core ----------------
// Wave computes a 64(M) x 64(N) tile: 4x4 accumulators of 16x16.
// NT layout: A is [M,K] row-major, B is [N,K] row-major (K contiguous for both).
// Per K=32 step: 4 A-frags + 4 B-frags (16x global_load_b128), 16 v_wmma.

__device__ __forceinline__ void mma_kspan(const bf16* __restrict__ A, int lda,
                                          const bf16* __restrict__ B, int ldb,
                                          int klen, v8f (&acc)[4][4]) {
    const int lane = threadIdx.x & 31;
    const int r    = lane & 15;
    const int kh   = (lane >> 4) * 16;
    for (int k0 = 0; k0 < klen; k0 += 32) {
        v16bf afrag[4];
#pragma unroll
        for (int mi = 0; mi < 4; ++mi)
            afrag[mi] = *(const v16bf*)(A + (size_t)(mi * 16 + r) * lda + k0 + kh);
        v16bf bfrag[4];
#pragma unroll
        for (int ni = 0; ni < 4; ++ni)
            bfrag[ni] = *(const v16bf*)(B + (size_t)(ni * 16 + r) * ldb + k0 + kh);
        if (k0 + 32 < klen) {  // pull next K-slabs toward L2 (global_prefetch)
            __builtin_prefetch(B + (size_t)r * ldb + k0 + 32 + kh, 0, 1);
            __builtin_prefetch(A + (size_t)r * lda + k0 + 32 + kh, 0, 1);
        }
#pragma unroll
        for (int mi = 0; mi < 4; ++mi)
#pragma unroll
            for (int ni = 0; ni < 4; ++ni)
                acc[mi][ni] = __builtin_amdgcn_wmma_f32_16x16x32_bf16(
                    false, afrag[mi], false, bfrag[ni], (short)0, acc[mi][ni],
                    false, false);
    }
}

// Block = 256 threads = 8 waves arranged 2(M) x 4(N) => block tile 128 x 256.
#define WAVE_ROW(wave) (((wave) >> 2) * 64)
#define WAVE_COL(wave) (((wave) & 3) * 64)

// ---------------- encode: Z0 = tanh(x @ Rw[a]^T + rb) ----------------

__global__ void __launch_bounds__(256) encode_gemm(
    const bf16* __restrict__ xbf,    // [BATCHN, DIN]
    const bf16* __restrict__ rwbf,   // [AREAS, AN, DIN]
    const float* __restrict__ rb,    // [AREAS, AN]
    float* __restrict__ hist0,       // [AREAS, BATCHN, AN]
    bf16* __restrict__ zbf)          // [AREAS, BATCHN, AN]
{
    const int a    = blockIdx.z;
    const int wave = threadIdx.x >> 5;
    const int lane = threadIdx.x & 31;
    const int rowBlk = blockIdx.y * 128 + WAVE_ROW(wave);  // batch rows
    const int colBlk = blockIdx.x * 256 + WAVE_COL(wave);  // neurons

    v8f zeroAcc = {};
    v8f acc[4][4];
#pragma unroll
    for (int mi = 0; mi < 4; ++mi)
#pragma unroll
        for (int ni = 0; ni < 4; ++ni) acc[mi][ni] = zeroAcc;

    mma_kspan(xbf + (size_t)rowBlk * DIN, DIN,
              rwbf + ((size_t)a * AN + colBlk) * DIN, DIN, DIN, acc);

    const int r  = lane & 15;
    const int hi = lane >> 4;
    float bias[4];
#pragma unroll
    for (int ni = 0; ni < 4; ++ni)
        bias[ni] = rb[(size_t)a * AN + colBlk + ni * 16 + r];
#pragma unroll
    for (int mi = 0; mi < 4; ++mi)
#pragma unroll
        for (int ni = 0; ni < 4; ++ni)
#pragma unroll
            for (int j = 0; j < 8; ++j) {
                int m = rowBlk + mi * 16 + j + hi * 8;
                int n = colBlk + ni * 16 + r;
                float v = tanhf(acc[mi][ni][j] + bias[ni]);
                size_t o = ((size_t)a * BATCHN + m) * AN + n;
                hist0[o] = v;
                zbf[o]   = (bf16)v;
            }
}

// ---------------- tick: Z = tanh(Σ_i gate[i]·Z[i]@Weff[o,i]^T + Σ_i gate[i]·b[o,i]) ----------------

__global__ void __launch_bounds__(256) tick_gemm(
    const bf16* __restrict__ zin,    // [AREAS, BATCHN, AN]
    const bf16* __restrict__ weff,   // [Ao, Ai, AN(out n), AN(in m)]
    const float* __restrict__ bblk,  // [Ao, Ai, AN]
    const float* __restrict__ gate,  // [AREAS]
    float* __restrict__ histT,       // [AREAS, BATCHN, AN]
    bf16* __restrict__ zout)         // [AREAS, BATCHN, AN]
{
    const int o    = blockIdx.z;
    const int wave = threadIdx.x >> 5;
    const int lane = threadIdx.x & 31;
    const int rowBlk = blockIdx.y * 128 + WAVE_ROW(wave);
    const int colBlk = blockIdx.x * 256 + WAVE_COL(wave);

    v8f zeroAcc = {};
    v8f acc[4][4];
#pragma unroll
    for (int mi = 0; mi < 4; ++mi)
#pragma unroll
        for (int ni = 0; ni < 4; ++ni) acc[mi][ni] = zeroAcc;

    for (int i = 0; i < AREAS; ++i) {
        if (gate[i] != 0.0f) {   // data-dependent skip (uniform branch)
            mma_kspan(zin + ((size_t)i * BATCHN + rowBlk) * AN, AN,
                      weff + (((size_t)o * AREAS + i) * AN + colBlk) * AN, AN,
                      AN, acc);
        }
    }

    const int r  = lane & 15;
    const int hi = lane >> 4;
    float bias[4];
#pragma unroll
    for (int ni = 0; ni < 4; ++ni) {
        int n = colBlk + ni * 16 + r;
        float s = 0.f;
        for (int i = 0; i < AREAS; ++i)
            s += gate[i] * bblk[((size_t)o * AREAS + i) * AN + n];
        bias[ni] = s;
    }
#pragma unroll
    for (int mi = 0; mi < 4; ++mi)
#pragma unroll
        for (int ni = 0; ni < 4; ++ni)
#pragma unroll
            for (int j = 0; j < 8; ++j) {
                int m = rowBlk + mi * 16 + j + hi * 8;
                int n = colBlk + ni * 16 + r;
                float v = tanhf(acc[mi][ni][j] + bias[ni]);
                size_t oo = ((size_t)o * BATCHN + m) * AN + n;
                histT[oo] = v;
                zout[oo]  = (bf16)v;
            }
}

// ---------------- output: logits = Zcat @ Owp^T + Ob ----------------

__global__ void __launch_bounds__(256) out_gemm(
    const bf16* __restrict__ zbf,    // [AREAS, BATCHN, AN]
    const bf16* __restrict__ owp,    // [NCLS, NTOT]
    const float* __restrict__ Ob,    // [NCLS]
    float* __restrict__ logits)      // [BATCHN, NCLS]
{
    const int wave = threadIdx.x >> 5;
    const int lane = threadIdx.x & 31;
    const int rowBlk = blockIdx.y * 128 + WAVE_ROW(wave);  // batch
    const int colBlk = blockIdx.x * 256 + WAVE_COL(wave);  // classes

    v8f zeroAcc = {};
    v8f acc[4][4];
#pragma unroll
    for (int mi = 0; mi < 4; ++mi)
#pragma unroll
        for (int ni = 0; ni < 4; ++ni) acc[mi][ni] = zeroAcc;

    for (int a = 0; a < AREAS; ++a) {
        mma_kspan(zbf + ((size_t)a * BATCHN + rowBlk) * AN, AN,
                  owp + (size_t)colBlk * NTOT + (size_t)a * AN, NTOT, AN, acc);
    }

    const int r  = lane & 15;
    const int hi = lane >> 4;
    float bias[4];
#pragma unroll
    for (int ni = 0; ni < 4; ++ni)
        bias[ni] = Ob[colBlk + ni * 16 + r];
#pragma unroll
    for (int mi = 0; mi < 4; ++mi)
#pragma unroll
        for (int ni = 0; ni < 4; ++ni)
#pragma unroll
            for (int j = 0; j < 8; ++j) {
                int m = rowBlk + mi * 16 + j + hi * 8;
                int n = colBlk + ni * 16 + r;
                logits[(size_t)m * NCLS + n] = acc[mi][ni][j] + bias[ni];
            }
}

// ---------------- host ----------------

extern "C" void kernel_launch(void* const* d_in, const int* in_sizes, int n_in,
                              void* d_out, int out_size, void* d_ws, size_t ws_size,
                              hipStream_t stream) {
    const float* x    = (const float*)d_in[0];
    const float* Rw   = (const float*)d_in[1];
    const float* rb   = (const float*)d_in[2];
    const float* Wb   = (const float*)d_in[3];
    const float* bb   = (const float*)d_in[4];
    const float* mask = (const float*)d_in[5];
    const float* Ow   = (const float*)d_in[6];
    const float* Ob   = (const float*)d_in[7];
    const int*   aidx = (const int*)d_in[8];

    float* out    = (float*)d_out;
    float* logits = out;
    float* hist   = out + (size_t)BATCHN * NCLS;               // [5, AREAS, BATCHN, AN]
    const size_t histSlot = (size_t)AREAS * BATCHN * AN;

    // workspace carve (~186 MB)
    char*  ws  = (char*)d_ws;
    size_t off = 0;
    auto take = [&](size_t bytes) -> void* {
        off = (off + 255) & ~(size_t)255;
        void* p = ws + off;
        off += bytes;
        return p;
    };
    bf16* weff = (bf16*)take((size_t)AREAS * AREAS * AN * AN * sizeof(bf16));
    bf16* xbf  = (bf16*)take((size_t)BATCHN * DIN * sizeof(bf16));
    bf16* rwbf = (bf16*)take((size_t)AREAS * AN * DIN * sizeof(bf16));
    bf16* owp  = (bf16*)take((size_t)NCLS * NTOT * sizeof(bf16));
    bf16* z0   = (bf16*)take((size_t)AREAS * BATCHN * AN * sizeof(bf16));
    bf16* z1   = (bf16*)take((size_t)AREAS * BATCHN * AN * sizeof(bf16));
    float* accb = (float*)take(AREAS * sizeof(float));
    float* gate = (float*)take(AREAS * sizeof(float));
    (void)ws_size; (void)in_sizes; (void)n_in; (void)out_size;

    // 1) preprocess to bf16 (one-time HBM streaming pass)
    {
        int n = AREAS * AREAS * AN * AN;                        // 67,108,864
        make_weff<<<dim3(n / 512), dim3(256), 0, stream>>>(Wb, mask, weff, n);
    }
    {
        int n = BATCHN * DIN;
        cvt_f32_bf16<<<dim3(n / 512), dim3(256), 0, stream>>>(x, xbf, n);
    }
    {
        int n = AREAS * AN * DIN;
        cvt_f32_bf16<<<dim3(n / 512), dim3(256), 0, stream>>>(Rw, rwbf, n);
    }
    make_owp<<<dim3(NTOT / 256, NCLS), dim3(256), 0, stream>>>(Ow, aidx, owp);

    // 2) encode -> history[0] + bf16 Z   (block tile 128M x 256N)
    encode_gemm<<<dim3(AN / 256, BATCHN / 128, AREAS), dim3(256), 0, stream>>>(
        xbf, rwbf, rb, hist, z0);

    // 3) recurrent ticks (Weff stays resident in L2 after tick 0)
    bf16* zbuf[2] = { z0, z1 };
    for (int t = 0; t < 4; ++t) {
        bf16* zcur = zbuf[t & 1];
        bf16* znxt = zbuf[(t + 1) & 1];
        zero_acc<<<dim3(1), dim3(32), 0, stream>>>(accb);
        abs_reduce<<<dim3(256, AREAS), dim3(256), 0, stream>>>(hist + (size_t)t * histSlot, accb);
        threshold_gate<<<dim3(1), dim3(32), 0, stream>>>(accb, gate);
        tick_gemm<<<dim3(AN / 256, BATCHN / 128, AREAS), dim3(256), 0, stream>>>(
            zcur, weff, bb, gate, hist + (size_t)(t + 1) * histSlot, znxt);
    }

    // 4) logits (final Z is in z0 after 4 ticks)
    out_gemm<<<dim3(NCLS / 256, BATCHN / 128, 1), dim3(256), 0, stream>>>(
        z0, owp, Ob, logits);
}
